// RnnSage_78675210928329
// MI455X (gfx1250) — matrix operand order repaired and checked by reading, not compiled, not run on hardware
//
#include <hip/hip_runtime.h>
#include <math.h>

typedef __attribute__((ext_vector_type(16))) _Float16 v16h;
typedef __attribute__((ext_vector_type(8)))  float    v8f;

#define T_STEPS 8
#define N_NODES 50000
#define E_EDGES 800000
#define B_BATCH 4096
#define F_DIM   128
#define H_DIM   64
#define SAGE_K  256   // [x(128) | mean(128)]
#define RNN_K   192   // [state(128) | h_prev(64)]
#define SAGE_S  (SAGE_K / 32)   // 8 K-steps
#define RNN_S   (RNN_K / 32)    // 6 K-steps

// ---------------------------------------------------------------------------
// Fragment layouts (CDNA5 v_wmma_f32_16x16x32_f16, wave32):
//  A 16x32 f16: lane m=lane&15, kb=(lane&16)?16:0; half h -> k=kb+h (contiguous, same as B)
//  B 32x16 f16: lane n=lane&15, kb=(lane&16)?16:0; half h -> k=kb+h (contiguous order chosen here, not AMD's table)
//  C/D 16x16 f32: lane n=lane&15, elem r -> M = r + ((lane&16)?8:0)
// We pre-pack A (LDS) and B (global) so each fragment is 16 contiguous halves
// per lane: packed index = ((step*TILES + ntile)*32 + lane)*16 + h.
// ---------------------------------------------------------------------------

// inverse map for A staging: element (row m, k) -> (step, lane, half)
__device__ __forceinline__ int a_slot(int m, int k) {
  int s  = k >> 5;
  int kk = k & 31;
  int b  = (kk >= 16) ? 1 : 0;
  int h  = kk & 15;
  int lane = m + (b << 4);
  return ((s << 5) + lane) * 16 + h;
}

__device__ __forceinline__ v8f wmma3(v16h ahi, v16h alo, v16h bhi, v16h blo, v8f c) {
  // 2-term Markidis split: hi*hi + hi*lo + lo*hi  (~f32 accuracy)
  c = __builtin_amdgcn_wmma_f32_16x16x32_f16(false, ahi, false, bhi, (short)0, c, false, false);
  c = __builtin_amdgcn_wmma_f32_16x16x32_f16(false, ahi, false, blo, (short)0, c, false, false);
  c = __builtin_amdgcn_wmma_f32_16x16x32_f16(false, alo, false, bhi, (short)0, c, false, false);
  return c;
}

// ---------------------------------------------------------------------------
// Weight pre-pack: stacked [W0 ; W1] (k_split rows from W0, rest from W1),
// emitted in B-fragment order, hi/lo f16 planes. Runs once per launch.
// ---------------------------------------------------------------------------
__global__ void k_pack(const float* __restrict__ W0, const float* __restrict__ W1,
                       int k_split, int steps,
                       _Float16* __restrict__ outHi, _Float16* __restrict__ outLo) {
  int i = blockIdx.x * blockDim.x + threadIdx.x;
  int total = steps * 4 * 32 * 16;
  if (i >= total) return;
  int h    = i & 15;
  int lane = (i >> 4) & 31;
  int nt   = (i >> 9) & 3;
  int s    = i >> 11;
  int n    = lane & 15;
  int kb   = (lane & 16) ? 16 : 0;
  int k    = s * 32 + kb + h;
  int col  = nt * 16 + n;
  float w = (k < k_split) ? W0[(size_t)k * H_DIM + col]
                          : W1[(size_t)(k - k_split) * H_DIM + col];
  _Float16 hi = (_Float16)w;
  outHi[i] = hi;
  outLo[i] = (_Float16)(w - (float)hi);
}

// ---------------------------------------------------------------------------
// Per-(t,graph) setup: inv = -1, agg_sel = 0, deg_sel = 0
// ---------------------------------------------------------------------------
__global__ void k_init(int* __restrict__ inv, float* __restrict__ agg,
                       float* __restrict__ deg) {
  int i = blockIdx.x * blockDim.x + threadIdx.x;
  if (i < N_NODES) inv[i] = -1;
  if (i < B_BATCH * F_DIM) agg[i] = 0.0f;
  if (i < B_BATCH) deg[i] = 0.0f;
}

__global__ void k_scatter(const int* __restrict__ idx_t, int* __restrict__ inv) {
  int b = blockIdx.x * blockDim.x + threadIdx.x;
  if (b < B_BATCH) inv[idx_t[b]] = b;   // duplicates: any winner; consumers re-read inv[node]
}

__global__ void k_zero(float* __restrict__ p, int n) {
  int i = blockIdx.x * blockDim.x + threadIdx.x;
  if (i < n) p[i] = 0.0f;
}

// ---------------------------------------------------------------------------
// Edge pass: thread-per-edge filter, ballot-compacted wave cooperation on the
// ~8% of edges whose dst is selected. 128 floats / edge, 4 per lane, f32 atomics.
// ---------------------------------------------------------------------------
__global__ __launch_bounds__(256) void k_edges(
    const int*   __restrict__ src, const int* __restrict__ dst,
    const int*   __restrict__ inv, const float* __restrict__ x,
    float*       __restrict__ agg, float* __restrict__ deg) {
  int e0   = blockIdx.x * blockDim.x + threadIdx.x;
  int lane = threadIdx.x & 31;
  if (e0 + 4096 < E_EDGES) {  // stream ahead: global_prefetch_b8
    __builtin_prefetch(dst + e0 + 4096, 0, 1);
    __builtin_prefetch(src + e0 + 4096, 0, 1);
  }
  int myslot = -1, mysrc = 0;
  if (e0 < E_EDGES) {
    int d = dst[e0];
    myslot = inv[d];
    if (myslot >= 0) mysrc = src[e0];
  }
  unsigned long long mask = __ballot(myslot >= 0);
  while (mask) {
    int l = __ffsll((long long)mask) - 1;
    mask &= mask - 1;
    int slot = __shfl(myslot, l);
    int s    = __shfl(mysrc, l);
    const float4 v = *(const float4*)(x + (size_t)s * F_DIM + lane * 4);
    float* a = agg + (size_t)slot * F_DIM + lane * 4;
    atomicAdd(a + 0, v.x); atomicAdd(a + 1, v.y);
    atomicAdd(a + 2, v.z); atomicAdd(a + 3, v.w);
    if (lane == 0) atomicAdd(deg + slot, 1.0f);
  }
}

// ---------------------------------------------------------------------------
// SAGE for a 16-row batch tile: [x | mean] @ [W_self ; W_neigh] + b, ReLU,
// row L2-normalize, scatter into states[t, :, g*H : (g+1)*H].
// Block: 128 threads = 4 waves; wave w owns H-columns [16w, 16w+16).
// ---------------------------------------------------------------------------
__global__ __launch_bounds__(128) void k_sage(
    const float* __restrict__ x,       // [N,F] for this (t,g)
    const int*   __restrict__ idx_t,   // [B]
    const int*   __restrict__ inv,
    const float* __restrict__ agg,     // [B,F]
    const float* __restrict__ deg,     // [B]
    const _Float16* __restrict__ WpHi, // packed B frags, hi plane
    const _Float16* __restrict__ WpLo, // packed B frags, lo plane
    const float* __restrict__ b_sage,  // [H]
    float*       __restrict__ states,  // [T,B,2H]
    int t, int g) {
  __shared__ alignas(32) _Float16 aHi[SAGE_S * 32 * 16];
  __shared__ alignas(32) _Float16 aLo[SAGE_S * 32 * 16];
  __shared__ float hTile[16][H_DIM];
  __shared__ float rowScale[16];
  __shared__ int   sNode[16];
  __shared__ int   sSlot[16];
  __shared__ float sInvDeg[16];

  const int tid = threadIdx.x;
  const int bt  = blockIdx.x;  // 16-row batch tile

  if (tid < 16) {
    int node = idx_t[bt * 16 + tid];
    int slot = inv[node];
    sNode[tid] = node;
    sSlot[tid] = slot;
    sInvDeg[tid] = 1.0f / fmaxf(deg[slot], 1.0f);
  }
  __syncthreads();

  // Stage A rows [x_row(128) | mean(128)] directly in fragment order, hi/lo split once.
  for (int i = tid; i < 16 * SAGE_K; i += 128) {
    int r = i >> 8, k = i & 255;
    float v = (k < F_DIM)
                  ? x[(size_t)sNode[r] * F_DIM + k]
                  : agg[(size_t)sSlot[r] * F_DIM + (k - F_DIM)] * sInvDeg[r];
    int o = a_slot(r, k);
    _Float16 hi = (_Float16)v;
    aHi[o] = hi;
    aLo[o] = (_Float16)(v - (float)hi);
  }
  __syncthreads();

  const int wv = tid >> 5, lane = tid & 31;
  const int ncol = wv * 16;
  const v16h* Bhi = (const v16h*)WpHi;
  const v16h* Blo = (const v16h*)WpLo;
  v8f c = {};
  for (int s = 0; s < SAGE_S; ++s) {
    v16h ahi = *(const v16h*)&aHi[((s << 5) + lane) * 16];
    v16h alo = *(const v16h*)&aLo[((s << 5) + lane) * 16];
    int bo = (s * 4 + wv) * 32 + lane;
    c = wmma3(ahi, alo, Bhi[bo], Blo[bo], c);
  }

  {  // bias + ReLU into LDS tile
    int n = lane & 15;
    int mb = (lane & 16) ? 8 : 0;
    float bias = b_sage[ncol + n];
#pragma unroll
    for (int r = 0; r < 8; ++r)
      hTile[mb + r][ncol + n] = fmaxf(c[r] + bias, 0.0f);
  }
  __syncthreads();

  if (tid < 16) {  // row L2 norm
    float ss = 0.0f;
    for (int j = 0; j < H_DIM; ++j) { float v = hTile[tid][j]; ss += v * v; }
    rowScale[tid] = 1.0f / fmaxf(sqrtf(ss), 1e-12f);
  }
  __syncthreads();

  for (int i = tid; i < 16 * H_DIM; i += 128) {
    int r = i >> 6, j = i & 63;
    size_t o = ((size_t)t * B_BATCH + bt * 16 + r) * (2 * H_DIM) + (size_t)g * H_DIM + j;
    states[o] = hTile[r][j] * rowScale[r];
  }
}

// ---------------------------------------------------------------------------
// One Elman RNN step: h' = relu([s | h] @ [W_ih ; W_hh] + b_ih + b_hh),
// out[t,b] = h' . W_out + b_out. Sequential over t via stream ordering.
// ---------------------------------------------------------------------------
__global__ __launch_bounds__(128) void k_rnn(
    const float* __restrict__ states,  // [T,B,2H]
    const float* __restrict__ hPrev,   // [B,H]
    const _Float16* __restrict__ WpHi, // packed [W_ih ; W_hh] B frags, hi
    const _Float16* __restrict__ WpLo, // packed lo
    const float* __restrict__ b_ih,    // [H]
    const float* __restrict__ b_hh,    // [H]
    const float* __restrict__ W_out,   // [H]
    const float* __restrict__ b_out,   // [1]
    float*       __restrict__ hNext,   // [B,H]
    float*       __restrict__ out,     // [T,B]
    int t) {
  __shared__ alignas(32) _Float16 aHi[RNN_S * 32 * 16];
  __shared__ alignas(32) _Float16 aLo[RNN_S * 32 * 16];
  __shared__ float hTile[16][H_DIM];

  const int tid = threadIdx.x;
  const int bt  = blockIdx.x;

  for (int i = tid; i < 16 * RNN_K; i += 128) {
    int r = i / RNN_K, k = i % RNN_K;
    int row = bt * 16 + r;
    float v = (k < 2 * H_DIM)
                  ? states[((size_t)t * B_BATCH + row) * (2 * H_DIM) + k]
                  : hPrev[(size_t)row * H_DIM + (k - 2 * H_DIM)];
    int o = a_slot(r, k);
    _Float16 hi = (_Float16)v;
    aHi[o] = hi;
    aLo[o] = (_Float16)(v - (float)hi);
  }
  __syncthreads();

  const int wv = tid >> 5, lane = tid & 31;
  const int ncol = wv * 16;
  const v16h* Bhi = (const v16h*)WpHi;
  const v16h* Blo = (const v16h*)WpLo;
  v8f c = {};
  for (int s = 0; s < RNN_S; ++s) {
    v16h ahi = *(const v16h*)&aHi[((s << 5) + lane) * 16];
    v16h alo = *(const v16h*)&aLo[((s << 5) + lane) * 16];
    int bo = (s * 4 + wv) * 32 + lane;
    c = wmma3(ahi, alo, Bhi[bo], Blo[bo], c);
  }

  {
    int n = lane & 15;
    int mb = (lane & 16) ? 8 : 0;
    float bias = b_ih[ncol + n] + b_hh[ncol + n];
#pragma unroll
    for (int r = 0; r < 8; ++r)
      hTile[mb + r][ncol + n] = fmaxf(c[r] + bias, 0.0f);
  }
  __syncthreads();

  for (int i = tid; i < 16 * H_DIM; i += 128) {
    int r = i >> 6, j = i & 63;
    hNext[(size_t)(bt * 16 + r) * H_DIM + j] = hTile[r][j];
  }
  if (tid < 16) {
    float acc = b_out[0];
    for (int j = 0; j < H_DIM; ++j) acc += hTile[tid][j] * W_out[j];
    out[(size_t)t * B_BATCH + bt * 16 + tid] = acc;
  }
}

// ---------------------------------------------------------------------------
extern "C" void kernel_launch(void* const* d_in, const int* in_sizes, int n_in,
                              void* d_out, int out_size, void* d_ws, size_t ws_size,
                              hipStream_t stream) {
  (void)in_sizes; (void)n_in; (void)out_size; (void)ws_size;
  const float* feats[2] = {(const float*)d_in[0], (const float*)d_in[1]};
  const int*   srcs[2]  = {(const int*)d_in[2], (const int*)d_in[4]};
  const int*   dsts[2]  = {(const int*)d_in[3], (const int*)d_in[5]};
  const int*   idxs[2]  = {(const int*)d_in[6], (const int*)d_in[7]};
  const float* W_self   = (const float*)d_in[8];
  const float* W_neigh  = (const float*)d_in[9];
  const float* b_sage   = (const float*)d_in[10];
  const float* W_ih     = (const float*)d_in[11];
  const float* b_ih     = (const float*)d_in[12];
  const float* W_hh     = (const float*)d_in[13];
  const float* b_hh     = (const float*)d_in[14];
  const float* W_out    = (const float*)d_in[15];
  const float* b_out    = (const float*)d_in[16];
  float*       out      = (float*)d_out;

  char* ws = (char*)d_ws;
  size_t off = 0;
  auto wsalloc = [&](size_t bytes) -> void* {
    void* p = ws + off;
    off += (bytes + 255) & ~(size_t)255;
    return p;
  };
  int*      inv     = (int*)wsalloc((size_t)N_NODES * 4);
  float*    agg     = (float*)wsalloc((size_t)B_BATCH * F_DIM * 4);
  float*    deg     = (float*)wsalloc((size_t)B_BATCH * 4);
  float*    states  = (float*)wsalloc((size_t)T_STEPS * B_BATCH * 2 * H_DIM * 4);
  float*    hA      = (float*)wsalloc((size_t)B_BATCH * H_DIM * 4);
  float*    hB      = (float*)wsalloc((size_t)B_BATCH * H_DIM * 4);
  const int sagePackN = SAGE_S * 4 * 32 * 16;  // 16384 halves
  const int rnnPackN  = RNN_S * 4 * 32 * 16;   // 12288 halves
  _Float16* sageWHi = (_Float16*)wsalloc((size_t)sagePackN * 2);
  _Float16* sageWLo = (_Float16*)wsalloc((size_t)sagePackN * 2);
  _Float16* rnnWHi  = (_Float16*)wsalloc((size_t)rnnPackN * 2);
  _Float16* rnnWLo  = (_Float16*)wsalloc((size_t)rnnPackN * 2);

  // Pack weights once (invariant across t,g); zero RNN h0.
  k_pack<<<(sagePackN + 255) / 256, 256, 0, stream>>>(W_self, W_neigh, F_DIM, SAGE_S,
                                                      sageWHi, sageWLo);
  k_pack<<<(rnnPackN + 255) / 256, 256, 0, stream>>>(W_ih, W_hh, 2 * H_DIM, RNN_S,
                                                     rnnWHi, rnnWLo);
  k_zero<<<(B_BATCH * H_DIM + 255) / 256, 256, 0, stream>>>(hA, B_BATCH * H_DIM);

  for (int t = 0; t < T_STEPS; ++t) {
    for (int g = 0; g < 2; ++g) {
      const float* x    = feats[g] + (size_t)t * N_NODES * F_DIM;
      const int*   srcT = srcs[g] + (size_t)t * E_EDGES;
      const int*   dstT = dsts[g] + (size_t)t * E_EDGES;
      const int*   idxT = idxs[g] + (size_t)t * B_BATCH;
      k_init<<<(B_BATCH * F_DIM + 255) / 256, 256, 0, stream>>>(inv, agg, deg);
      k_scatter<<<(B_BATCH + 255) / 256, 256, 0, stream>>>(idxT, inv);
      k_edges<<<(E_EDGES + 255) / 256, 256, 0, stream>>>(srcT, dstT, inv, x, agg, deg);
      k_sage<<<B_BATCH / 16, 128, 0, stream>>>(x, idxT, inv, agg, deg,
                                               sageWHi, sageWLo, b_sage, states, t, g);
    }
  }

  float* hp = hA; float* hn = hB;
  for (int t = 0; t < T_STEPS; ++t) {
    k_rnn<<<B_BATCH / 16, 128, 0, stream>>>(states, hp, rnnWHi, rnnWLo,
                                            b_ih, b_hh, W_out, b_out, hn, out, t);
    float* tmp = hp; hp = hn; hn = tmp;
  }
}
